// DecoderWithAttention_59863254171904
// MI455X (gfx1250) — matrix operand order repaired and checked by reading, not compiled, not run on hardware
//
#include <hip/hip_runtime.h>

// ---------------------------------------------------------------------------
// Show-Attend-Tell decoder forward for MI455X (gfx1250), wave32 + WMMA bf16.
// B=128 batch rows -> 8 persistent blocks of 16 rows; all per-step GEMMs are
// M=16 tiles computed with v_wmma_f32_16x16x32_bf16 (fp32 accumulate).
// Hidden state lives in LDS (async-staged); the big fc GEMM is hoisted out of
// the recurrence and runs once over all (b,t) rows at full occupancy.
// ---------------------------------------------------------------------------

constexpr int BB  = 128;    // batch
constexpr int PP  = 196;    // pixels
constexpr int ENC = 2048;   // encoder dim
constexpr int LL  = 50;     // max caption length
constexpr int VV  = 10000;  // vocab
constexpr int AA  = 512;    // attention dim
constexpr int DD  = 512;    // decoder dim
constexpr int EE  = 512;    // embed dim
constexpr int TT  = LL - 1; // 49 decode steps

typedef __attribute__((ext_vector_type(16))) __bf16 v16bf;
typedef __attribute__((ext_vector_type(8)))  float  v8f;

struct alignas(16) U16x8 { unsigned short x[8]; };
union Frag { v16bf v; unsigned short u[16]; U16x8 h[2]; };
union Acc  { v8f v; float f[8]; };

static __device__ __forceinline__ unsigned short f2bf(float f) {
  unsigned u = __float_as_uint(f);
  unsigned r = u + 0x7FFFu + ((u >> 16) & 1u);   // round-to-nearest-even
  return (unsigned short)(r >> 16);
}
static __device__ __forceinline__ float bf2f(unsigned short h) {
  return __uint_as_float(((unsigned)h) << 16);
}
static __device__ __forceinline__ float sigm(float x) {
  return 1.0f / (1.0f + __expf(-x));
}

// A fragment: 16x32 bf16 tile from row-major activations (global or LDS).
// lane: row m = lane&15, hi = lane>>4; e0..7 -> K=k0+8*hi.., e8..15 -> +16.
static __device__ __forceinline__ v16bf load_a_bf16(const unsigned short* base,
                                                    int ld, int k0, int lane) {
  int m = lane & 15, hi = lane >> 4;
  const unsigned short* p = base + (size_t)m * ld + k0 + 8 * hi;
  Frag f;
  f.h[0] = *(const U16x8*)(p);
  f.h[1] = *(const U16x8*)(p + 16);
  return f.v;
}

// B fragment: pre-packed fragment-major weights (16 contiguous bf16 per lane).
static __device__ __forceinline__ v16bf load_b_pack(const unsigned short* pack,
                                                    int ntiles, int kt, int nt,
                                                    int lane) {
  const unsigned short* p = pack + (((size_t)kt * ntiles + nt) * 32 + lane) * 16;
  Frag f;
  f.h[0] = *(const U16x8*)(p);
  f.h[1] = *(const U16x8*)(p + 8);
  return f.v;
}

static __device__ __forceinline__ v8f wmma_bf16(v16bf a, v16bf b, v8f c) {
  return __builtin_amdgcn_wmma_f32_16x16x32_bf16(false, a, false, b,
                                                 (short)0, c, false, false);
}

// Async copy of 32 bytes global -> LDS per calling lane (2 x B128), ASYNCcnt.
static __device__ __forceinline__ void async_g2l_32B(unsigned lds_addr,
                                                     unsigned long long gaddr) {
  asm volatile("global_load_async_to_lds_b128 %0, %1, off"
               :: "v"(lds_addr), "v"(gaddr) : "memory");
  asm volatile("global_load_async_to_lds_b128 %0, %1, off offset:16"
               :: "v"(lds_addr), "v"(gaddr) : "memory");
}
static __device__ __forceinline__ void wait_async0() {
  asm volatile("s_wait_asynccnt 0x0" ::: "memory");
}
// Numeric LDS offset = low 32 bits of the generic pointer (LDS aperture).
static __device__ __forceinline__ unsigned lds_off(const void* p) {
  return (unsigned)(unsigned long long)p;
}

// ---------------------------------------------------------------------------
// K0: stable descending sort of cap_len + caption gather + small outputs.
// ---------------------------------------------------------------------------
__global__ void sort_prep_kernel(const int* __restrict__ cap_len,
                                 const int* __restrict__ enc_cap,
                                 int* __restrict__ sort_ind,
                                 int* __restrict__ dec_len_s,
                                 int* __restrict__ caps_s,
                                 float* __restrict__ out_caps,
                                 float* __restrict__ out_declen,
                                 float* __restrict__ out_sortind) {
  __shared__ int s_cap[BB];
  __shared__ int s_ind[BB];
  int i = threadIdx.x;
  s_cap[i] = cap_len[i];
  __syncthreads();
  int key = s_cap[i];
  int rank = 0;
  for (int j = 0; j < BB; ++j) {
    int kj = s_cap[j];
    if (kj > key || (kj == key && j < i)) ++rank;   // stable, descending
  }
  s_ind[rank] = i;
  __syncthreads();
  int src = s_ind[i];
  sort_ind[i] = src;
  int dl = s_cap[src] - 1;
  dec_len_s[i] = dl;
  out_declen[i] = (float)dl;
  out_sortind[i] = (float)src;
  for (int t = 0; t < LL; ++t) {
    int cv = enc_cap[src * LL + t];
    caps_s[i * LL + t] = cv;
    out_caps[i * LL + t] = (float)cv;
  }
}

// ---------------------------------------------------------------------------
// K1: pack a weight W[N][K] (used as y = x @ W^T) into WMMA-B fragment layout.
// ---------------------------------------------------------------------------
__global__ void pack_weight_kernel(const float* __restrict__ src,
                                   unsigned short* __restrict__ dst,
                                   int N, int K) {
  int ntiles = N >> 4, ktiles = K >> 5;
  long long total = (long long)ntiles * ktiles * 32;
  for (long long idx = (long long)blockIdx.x * blockDim.x + threadIdx.x;
       idx < total; idx += (long long)gridDim.x * blockDim.x) {
    int lane = (int)(idx & 31);
    long long tile = idx >> 5;
    int nt = (int)(tile % ntiles);
    int kt = (int)(tile / ntiles);
    int n  = nt * 16 + (lane & 15);
    int kb = kt * 32 + (lane >> 4) * 8;
    const float* s = src + (size_t)n * K + kb;
    unsigned short* d = dst + (((size_t)kt * ntiles + nt) * 32 + lane) * 16;
#pragma unroll
    for (int e = 0; e < 16; ++e) d[e] = f2bf(s[e < 8 ? e : e + 8]);
  }
}

// ---------------------------------------------------------------------------
// K2a: gathered encoder mean -> bf16 [BB][ENC]
// ---------------------------------------------------------------------------
__global__ void enc_mean_kernel(const float* __restrict__ enc,
                                const int* __restrict__ sind,
                                unsigned short* __restrict__ mean_b) {
  int idx = blockIdx.x * blockDim.x + threadIdx.x;
  if (idx >= BB * ENC) return;
  int b = idx >> 11, e = idx & (ENC - 1);
  const float* src = enc + ((size_t)sind[b] * PP) * ENC + e;
  float s = 0.f;
  for (int p = 0; p < PP; ++p) s += src[(size_t)p * ENC];
  mean_b[idx] = f2bf(s * (1.0f / PP));
}

// ---------------------------------------------------------------------------
// K2b: embedding gather -> bf16 [BB][LL][EE]
// ---------------------------------------------------------------------------
__global__ void emb_gather_kernel(const float* __restrict__ emb_W,
                                  const int* __restrict__ caps_s,
                                  unsigned short* __restrict__ emb_b) {
  int idx = blockIdx.x * blockDim.x + threadIdx.x;
  if (idx >= BB * LL * EE) return;
  int e  = idx & (EE - 1);
  int bt = idx >> 9;
  int tok = caps_s[bt];
  emb_b[idx] = f2bf(emb_W[(size_t)tok * EE + e]);
}

// ---------------------------------------------------------------------------
// K2c: sorted encoder copy -> bf16 [BB][PP][ENC] (L2-resident ctx operand).
// ---------------------------------------------------------------------------
__global__ void enc_gather_bf16_kernel(const float* __restrict__ enc,
                                       const int* __restrict__ sind,
                                       unsigned short* __restrict__ enc_sb) {
  long long idx4 = (long long)blockIdx.x * blockDim.x + threadIdx.x;
  long long total4 = (long long)BB * PP * ENC / 4;
  if (idx4 >= total4) return;
  long long o = idx4 * 4;
  int e  = (int)(o & (ENC - 1));
  long long bp = o >> 11;                 // sorted (b*PP+p)
  int b = (int)(bp / PP), p = (int)(bp % PP);
  const float* s = enc + ((size_t)sind[b] * PP + p) * ENC + e;
  unsigned short r[4];
#pragma unroll
  for (int q = 0; q < 4; ++q) r[q] = f2bf(s[q]);
  *(unsigned long long*)(enc_sb + o) =
      (unsigned long long)r[0] | ((unsigned long long)r[1] << 16) |
      ((unsigned long long)r[2] << 32) | ((unsigned long long)r[3] << 48);
}

// ---------------------------------------------------------------------------
// K3: enc_feat = gathered(enc) @ enc_attn_W^T + b  -> bf16 [BB*PP][AA]
// ---------------------------------------------------------------------------
__global__ void enc_feat_kernel(const float* __restrict__ enc,
                                const int* __restrict__ sind,
                                const unsigned short* __restrict__ wenc,
                                const float* __restrict__ enc_attn_b,
                                unsigned short* __restrict__ enc_featb) {
  int wid  = blockIdx.x * (blockDim.x >> 5) + (threadIdx.x >> 5);
  int lane = threadIdx.x & 31;
  const int mtiles = (BB * PP) / 16;      // 1568
  const int ntiles = AA / 16;             // 32
  if (wid >= mtiles * ntiles) return;
  int mt = wid / ntiles;
  int nt = wid % ntiles;
  int m = lane & 15, hi = lane >> 4;
  int g = mt * 16 + m;
  int b = g / PP, p = g % PP;
  const float* arow = enc + ((size_t)sind[b] * PP + p) * ENC;
  Acc acc;
#pragma unroll
  for (int z = 0; z < 8; ++z) acc.f[z] = 0.f;
  for (int kt = 0; kt < ENC / 32; ++kt) {
    int kb = kt * 32 + 8 * hi;
    Frag a;
#pragma unroll
    for (int e = 0; e < 8; ++e) a.u[e] = f2bf(arow[kb + e]);
#pragma unroll
    for (int e = 0; e < 8; ++e) a.u[8 + e] = f2bf(arow[kb + 16 + e]);
    v16bf bf = load_b_pack(wenc, ntiles, kt, nt, lane);
    acc.v = wmma_bf16(a.v, bf, acc.v);
  }
  int n = nt * 16 + (lane & 15);
  float bias = enc_attn_b[n];
#pragma unroll
  for (int v = 0; v < 8; ++v) {
    int gg = mt * 16 + 8 * hi + v;
    enc_featb[(size_t)gg * AA + n] = f2bf(acc.f[v] + bias);
  }
}

// ---------------------------------------------------------------------------
// K4: h0/c0 = enc_mean @ init_{h,c}_W^T + b
// ---------------------------------------------------------------------------
__global__ void init_state_kernel(const unsigned short* __restrict__ mean_b,
                                  const unsigned short* __restrict__ winh,
                                  const unsigned short* __restrict__ winc,
                                  const float* __restrict__ hb,
                                  const float* __restrict__ cb,
                                  float* __restrict__ h_f,
                                  float* __restrict__ c_f,
                                  unsigned short* __restrict__ h_b) {
  int wid  = blockIdx.x * (blockDim.x >> 5) + (threadIdx.x >> 5);
  int lane = threadIdx.x & 31;
  if (wid >= (BB / 16) * (DD / 16) * 2) return;   // 512 wave-tiles
  int which = wid & 1;
  int nt = (wid >> 1) & 31;
  int mt = wid >> 6;
  const unsigned short* wp = which ? winc : winh;
  int hi = lane >> 4;
  Acc acc;
#pragma unroll
  for (int z = 0; z < 8; ++z) acc.f[z] = 0.f;
  const unsigned short* abase = mean_b + (size_t)mt * 16 * ENC;
  for (int kt = 0; kt < ENC / 32; ++kt) {
    v16bf a = load_a_bf16(abase, ENC, kt * 32, lane);
    v16bf b = load_b_pack(wp, DD / 16, kt, nt, lane);
    acc.v = wmma_bf16(a, b, acc.v);
  }
  int n = nt * 16 + (lane & 15);
  float bias = which ? cb[n] : hb[n];
#pragma unroll
  for (int v = 0; v < 8; ++v) {
    int row = mt * 16 + 8 * hi + v;
    float val = acc.f[v] + bias;
    if (which) {
      c_f[(size_t)row * DD + n] = val;
    } else {
      h_f[(size_t)row * DD + n] = val;
      h_b[(size_t)row * DD + n] = f2bf(val);
    }
  }
}

// ---------------------------------------------------------------------------
// K5: persistent decoder (recurrence only; fc GEMM hoisted out).
// 8 blocks x 512 threads (16 waves); block owns 16 batch rows.
// ---------------------------------------------------------------------------
__global__ void __launch_bounds__(512)
decoder_step_kernel(const float* __restrict__ enc,
                    const unsigned short* __restrict__ enc_sb,  // may be unused
                    int use_bf16_enc,
                    const int* __restrict__ sind,
                    const int* __restrict__ dec_len_s,
                    const unsigned short* __restrict__ enc_featb,
                    const unsigned short* __restrict__ emb_b,
                    const unsigned short* __restrict__ wdec,
                    const float* __restrict__ dec_bias,
                    const float* __restrict__ wfull,
                    const float* __restrict__ full_bias,
                    const unsigned short* __restrict__ wfbeta,
                    const float* __restrict__ fbeta_bias,
                    const unsigned short* __restrict__ wih,
                    const unsigned short* __restrict__ whh,
                    const float* __restrict__ bih,
                    const float* __restrict__ bhh,
                    float* __restrict__ h_f, float* __restrict__ c_f,
                    unsigned short* __restrict__ h_b,
                    float* __restrict__ ctx_f,
                    unsigned short* __restrict__ xctx_b,
                    float* __restrict__ gates_f,
                    unsigned short* __restrict__ h_hist,   // [BB*TT][DD] bf16
                    float* __restrict__ out_alphas) {
  __shared__ alignas(16) float          s_dec[16][AA];    // 32 KB (reused for emb)
  __shared__ alignas(16) float          s_alpha[16][PP];  // 12.25 KB
  __shared__ alignas(16) unsigned short s_hb[16][DD];     // 16 KB hidden state

  int tid  = threadIdx.x;
  int wave = tid >> 5, lane = tid & 31;
  int hi   = lane >> 4;
  int r0   = blockIdx.x * 16;

  unsigned short* s_emb = (unsigned short*)&s_dec[0][0];  // 16 KB alias

  // Prologue: async-stage h0 (16x512 bf16 = 16 KB) global -> LDS.
  {
    unsigned la = lds_off(&s_hb[0][0]) + (unsigned)tid * 32u;
    unsigned long long ga =
        (unsigned long long)(const void*)(h_b + (size_t)r0 * DD) +
        (unsigned long long)tid * 32ull;
    async_g2l_32B(la, ga);
    wait_async0();
  }
  __syncthreads();

  for (int t = 0; t < TT; ++t) {
    // ---- (a) dec_feat = h @ dec_attn_W^T + b ------------------------------
    for (int nt = wave; nt < AA / 16; nt += 16) {
      Acc acc;
#pragma unroll
      for (int z = 0; z < 8; ++z) acc.f[z] = 0.f;
      for (int kt = 0; kt < DD / 32; ++kt) {
        v16bf a = load_a_bf16(&s_hb[0][0], DD, kt * 32, lane);
        v16bf b = load_b_pack(wdec, AA / 16, kt, nt, lane);
        acc.v = wmma_bf16(a, b, acc.v);
      }
      int n = nt * 16 + (lane & 15);
      float bias = dec_bias[n];
#pragma unroll
      for (int v = 0; v < 8; ++v) s_dec[8 * hi + v][n] = acc.f[v] + bias;
    }
    __syncthreads();

    // ---- (b) scores + softmax: wave w owns local batch row w --------------
    {
      int mb = wave;
      int gb = r0 + mb;
      float act = (dec_len_s[gb] > t) ? 1.0f : 0.0f;
      const unsigned short* efrow = enc_featb + ((size_t)gb * PP) * AA;
      float sc[7];
      int cnt = 0;
      float mx = -3.4e38f;
      for (int p = lane; p < PP; p += 32) {
        const unsigned short* er = efrow + (size_t)p * AA;
        float s = full_bias[0];
        for (int a = 0; a < AA; a += 8) {
          U16x8 ch = *(const U16x8*)(er + a);
#pragma unroll
          for (int q = 0; q < 8; ++q) {
            float vv = bf2f(ch.x[q]) + s_dec[mb][a + q];
            vv = vv > 0.f ? vv : 0.f;             // ReLU
            s += vv * wfull[a + q];
          }
        }
        sc[cnt++] = s;
        mx = fmaxf(mx, s);
      }
#pragma unroll
      for (int off = 16; off; off >>= 1) mx = fmaxf(mx, __shfl_xor(mx, off, 32));
      float sum = 0.f;
      for (int i2 = 0; i2 < cnt; ++i2) { sc[i2] = __expf(sc[i2] - mx); sum += sc[i2]; }
#pragma unroll
      for (int off = 16; off; off >>= 1) sum += __shfl_xor(sum, off, 32);
      float inv = 1.0f / sum;
      cnt = 0;
      for (int p = lane; p < PP; p += 32) {
        float al = sc[cnt++] * inv;
        s_alpha[mb][p] = al;                               // unmasked for ctx
        out_alphas[((size_t)gb * TT + t) * PP + p] = al * act;  // masked out
      }
    }
    __syncthreads();

    // Kick async stage of emb[:, t, :] (16 x 512 bf16) into s_emb (s_dec alias).
    {
      int row = tid >> 5;                 // wave w stages row w
      unsigned colb = (unsigned)(lane) * 32u;
      unsigned la = lds_off(s_emb) + (unsigned)row * 1024u + colb;
      unsigned long long ga =
          (unsigned long long)(const void*)(emb_b +
              ((size_t)(r0 + row) * LL + t) * EE) + colb;
      async_g2l_32B(la, ga);
    }

    // ---- (c) ctx[b,e] = sum_p alpha * enc_s -------------------------------
    for (int idx = tid; idx < 16 * ENC; idx += 512) {
      int mb = idx >> 11, e = idx & (ENC - 1);
      int gb = r0 + mb;
      float s = 0.f;
      if (use_bf16_enc) {
        const unsigned short* srow = enc_sb + ((size_t)gb * PP) * ENC + e;
        for (int p = 0; p < PP; ++p) s += s_alpha[mb][p] * bf2f(srow[(size_t)p * ENC]);
      } else {
        const float* srow = enc + ((size_t)sind[gb] * PP) * ENC + e;
        for (int p = 0; p < PP; ++p) {
          if ((p & 31) == 0) __builtin_prefetch(srow + (size_t)(p + 32) * ENC, 0, 0);
          s += s_alpha[mb][p] * srow[(size_t)p * ENC];
        }
      }
      ctx_f[(size_t)gb * ENC + e] = s;
    }
    __threadfence_block();
    __syncthreads();

    // ---- (d) gate = sigmoid(h @ fbeta^T + b); xctx = bf16(gate * ctx) -----
    for (int nt = wave; nt < ENC / 16; nt += 16) {
      Acc acc;
#pragma unroll
      for (int z = 0; z < 8; ++z) acc.f[z] = 0.f;
      for (int kt = 0; kt < DD / 32; ++kt) {
        v16bf a = load_a_bf16(&s_hb[0][0], DD, kt * 32, lane);
        v16bf b = load_b_pack(wfbeta, ENC / 16, kt, nt, lane);
        acc.v = wmma_bf16(a, b, acc.v);
      }
      int n = nt * 16 + (lane & 15);
      float bias = fbeta_bias[n];
#pragma unroll
      for (int v = 0; v < 8; ++v) {
        int gb = r0 + 8 * hi + v;
        float g  = sigm(acc.f[v] + bias);
        float xv = g * ctx_f[(size_t)gb * ENC + n];
        xctx_b[(size_t)gb * ENC + n] = f2bf(xv);
      }
    }
    __threadfence_block();
    wait_async0();                         // emb slice landed in s_emb
    __syncthreads();

    // ---- (e) gates = [emb|xctx] @ Wih^T + h @ Whh^T + bih + bhh -----------
    for (int nt = wave; nt < 128; nt += 16) {
      Acc acc;
#pragma unroll
      for (int z = 0; z < 8; ++z) acc.f[z] = 0.f;
      for (int kt = 0; kt < EE / 32; ++kt) {               // K = 0..511 (emb, LDS)
        v16bf a = load_a_bf16(s_emb, EE, kt * 32, lane);
        v16bf b = load_b_pack(wih, 128, kt, nt, lane);
        acc.v = wmma_bf16(a, b, acc.v);
      }
      for (int kt = 16; kt < 80; ++kt) {                   // K = 512..2559 (ctx)
        v16bf a = load_a_bf16(xctx_b + (size_t)r0 * ENC, ENC, (kt - 16) * 32, lane);
        v16bf b = load_b_pack(wih, 128, kt, nt, lane);
        acc.v = wmma_bf16(a, b, acc.v);
      }
      for (int kt = 0; kt < DD / 32; ++kt) {               // + h @ Whh^T (LDS)
        v16bf a = load_a_bf16(&s_hb[0][0], DD, kt * 32, lane);
        v16bf b = load_b_pack(whh, 128, kt, nt, lane);
        acc.v = wmma_bf16(a, b, acc.v);
      }
      int n = nt * 16 + (lane & 15);
      float bias = bih[n] + bhh[n];
#pragma unroll
      for (int v = 0; v < 8; ++v) {
        int gb = r0 + 8 * hi + v;
        gates_f[(size_t)gb * (4 * DD) + n] = acc.f[v] + bias;
      }
    }
    __threadfence_block();
    __syncthreads();

    // ---- (f) LSTM pointwise + state freeze; log h into history ------------
    for (int idx = tid; idx < 16 * DD; idx += 512) {
      int mb = idx >> 9, j = idx & (DD - 1);
      int gb = r0 + mb;
      const float* gr = gates_f + (size_t)gb * (4 * DD);
      float ig = sigm(gr[j]);
      float fg = sigm(gr[DD + j]);
      float gg = tanhf(gr[2 * DD + j]);
      float og = sigm(gr[3 * DD + j]);
      float co = c_f[(size_t)gb * DD + j];
      float cn = fg * co + ig * gg;
      float hn = og * tanhf(cn);
      bool act = dec_len_s[gb] > t;
      float ho = h_f[(size_t)gb * DD + j];
      float hs = act ? hn : ho;
      float cs = act ? cn : co;
      h_f[(size_t)gb * DD + j] = hs;
      c_f[(size_t)gb * DD + j] = cs;
      unsigned short hb16 = f2bf(hs);
      s_hb[mb][j] = hb16;
      h_hist[((size_t)gb * TT + t) * DD + j] = hb16;   // for hoisted fc GEMM
    }
    __threadfence_block();
    __syncthreads();
  }
}

// ---------------------------------------------------------------------------
// K6: hoisted fc GEMM over ALL (b,t) rows at once:
// logits[B*T, V] = h_hist[B*T, D] @ fc_W^T + fc_b, masked by active(b,t).
// 392 x 625 = 245,000 wave-tiles -> full-GPU WMMA occupancy.
// ---------------------------------------------------------------------------
__global__ void fc_logits_kernel(const unsigned short* __restrict__ h_hist,
                                 const unsigned short* __restrict__ wfc,
                                 const float* __restrict__ fc_bias,
                                 const int* __restrict__ dec_len_s,
                                 float* __restrict__ out_logits) {
  int wid  = blockIdx.x * (blockDim.x >> 5) + (threadIdx.x >> 5);
  int lane = threadIdx.x & 31;
  const int mtiles = (BB * TT) / 16;   // 392
  const int ntiles = VV / 16;          // 625
  if (wid >= mtiles * ntiles) return;
  int mt = wid / ntiles;
  int nt = wid % ntiles;
  int hi = lane >> 4;
  Acc acc;
#pragma unroll
  for (int z = 0; z < 8; ++z) acc.f[z] = 0.f;
  const unsigned short* abase = h_hist + (size_t)mt * 16 * DD;
  for (int kt = 0; kt < DD / 32; ++kt) {
    v16bf a = load_a_bf16(abase, DD, kt * 32, lane);
    v16bf b = load_b_pack(wfc, ntiles, kt, nt, lane);
    acc.v = wmma_bf16(a, b, acc.v);
  }
  int n = nt * 16 + (lane & 15);
  float bias = fc_bias[n];
#pragma unroll
  for (int v = 0; v < 8; ++v) {
    int g  = mt * 16 + 8 * hi + v;        // flattened (gb*TT + t)
    int gb = g / TT, t = g % TT;
    float act = (dec_len_s[gb] > t) ? 1.0f : 0.0f;
    out_logits[(size_t)g * VV + n] = (acc.f[v] + bias) * act;
  }
}

// ---------------------------------------------------------------------------
// Host launcher
// ---------------------------------------------------------------------------
extern "C" void kernel_launch(void* const* d_in, const int* in_sizes, int n_in,
                              void* d_out, int out_size, void* d_ws, size_t ws_size,
                              hipStream_t stream) {
  (void)in_sizes; (void)n_in; (void)out_size;

  const float* encoder_out = (const float*)d_in[0];
  const int*   enc_cap     = (const int*)d_in[1];
  const int*   cap_len     = (const int*)d_in[2];
  const float* emb_W       = (const float*)d_in[3];
  const float* init_h_W    = (const float*)d_in[4];
  const float* init_h_b    = (const float*)d_in[5];
  const float* init_c_W    = (const float*)d_in[6];
  const float* init_c_b    = (const float*)d_in[7];
  const float* enc_attn_W  = (const float*)d_in[8];
  const float* enc_attn_b  = (const float*)d_in[9];
  const float* dec_attn_W  = (const float*)d_in[10];
  const float* dec_attn_b  = (const float*)d_in[11];
  const float* full_attn_W = (const float*)d_in[12];
  const float* full_attn_b = (const float*)d_in[13];
  const float* lstm_Wih    = (const float*)d_in[14];
  const float* lstm_Whh    = (const float*)d_in[15];
  const float* lstm_bih    = (const float*)d_in[16];
  const float* lstm_bhh    = (const float*)d_in[17];
  const float* fbeta_W     = (const float*)d_in[18];
  const float* fbeta_b     = (const float*)d_in[19];
  const float* fc_W        = (const float*)d_in[20];
  const float* fc_b        = (const float*)d_in[21];

  // Output tuple: (output[B,T,V], caps_s[B,L], dec_len[B], alphas[B,T,P], sort_ind[B])
  float* out         = (float*)d_out;
  float* out_logits  = out;
  float* out_caps    = out_logits + (size_t)BB * TT * VV;
  float* out_declen  = out_caps + (size_t)BB * LL;
  float* out_alphas  = out_declen + BB;
  float* out_sortind = out_alphas + (size_t)BB * TT * PP;

  // Workspace carve-out, 256B aligned.
  char* w = (char*)d_ws;
  auto alloc = [&](size_t bytes) -> void* {
    char* p = w;
    w += (bytes + 255) & ~(size_t)255;
    return p;
  };
  unsigned short* wdec      = (unsigned short*)alloc((size_t)DD * AA * 2);
  unsigned short* wfbeta    = (unsigned short*)alloc((size_t)DD * ENC * 2);
  unsigned short* wih       = (unsigned short*)alloc((size_t)(EE + ENC) * (4 * DD) * 2);
  unsigned short* whh       = (unsigned short*)alloc((size_t)DD * (4 * DD) * 2);
  unsigned short* wfc       = (unsigned short*)alloc((size_t)DD * VV * 2);
  unsigned short* winh      = (unsigned short*)alloc((size_t)ENC * DD * 2);
  unsigned short* winc      = (unsigned short*)alloc((size_t)ENC * DD * 2);
  unsigned short* wenc      = (unsigned short*)alloc((size_t)ENC * AA * 2);
  unsigned short* enc_featb = (unsigned short*)alloc((size_t)BB * PP * AA * 2);
  unsigned short* mean_b    = (unsigned short*)alloc((size_t)BB * ENC * 2);
  unsigned short* emb_b     = (unsigned short*)alloc((size_t)BB * LL * EE * 2);
  unsigned short* h_b       = (unsigned short*)alloc((size_t)BB * DD * 2);
  unsigned short* h_hist    = (unsigned short*)alloc((size_t)BB * TT * DD * 2);
  float*          h_f       = (float*)alloc((size_t)BB * DD * 4);
  float*          c_f       = (float*)alloc((size_t)BB * DD * 4);
  float*          ctx_f     = (float*)alloc((size_t)BB * ENC * 4);
  unsigned short* xctx_b    = (unsigned short*)alloc((size_t)BB * ENC * 2);
  float*          gates_f   = (float*)alloc((size_t)BB * (4 * DD) * 4);
  int*            sort_ind  = (int*)alloc(BB * 4);
  int*            dec_len_s = (int*)alloc(BB * 4);
  int*            caps_s    = (int*)alloc((size_t)BB * LL * 4);

  // Optional: sorted bf16 encoder copy (103 MB) if workspace allows.
  size_t used = (size_t)(w - (char*)d_ws);
  size_t enc_sb_bytes = (size_t)BB * PP * ENC * 2;
  int use_bf16_enc = (used + enc_sb_bytes <= ws_size) ? 1 : 0;
  unsigned short* enc_sb = (unsigned short*)alloc(enc_sb_bytes);

  // K0: sort + caption gather
  sort_prep_kernel<<<1, BB, 0, stream>>>(cap_len, enc_cap, sort_ind, dec_len_s,
                                         caps_s, out_caps, out_declen, out_sortind);

  // K1: pack all weights into WMMA-B fragment layout (bf16)
  auto pack = [&](const float* src, unsigned short* dst, int N, int K) {
    long long total = (long long)(N / 16) * (K / 32) * 32;
    int blocks = (int)((total + 255) / 256);
    pack_weight_kernel<<<blocks, 256, 0, stream>>>(src, dst, N, K);
  };
  pack(dec_attn_W, wdec, AA, DD);
  pack(fbeta_W,    wfbeta, ENC, DD);
  pack(lstm_Wih,   wih, 4 * DD, EE + ENC);
  pack(lstm_Whh,   whh, 4 * DD, DD);
  pack(fc_W,       wfc, VV, DD);
  pack(init_h_W,   winh, DD, ENC);
  pack(init_c_W,   winc, DD, ENC);
  pack(enc_attn_W, wenc, AA, ENC);

  // K2: encoder mean + embedding gather (+ optional bf16 encoder copy)
  enc_mean_kernel<<<(BB * ENC + 255) / 256, 256, 0, stream>>>(encoder_out, sort_ind, mean_b);
  emb_gather_kernel<<<(BB * LL * EE + 255) / 256, 256, 0, stream>>>(emb_W, caps_s, emb_b);
  if (use_bf16_enc) {
    long long tot4 = (long long)BB * PP * ENC / 4;
    enc_gather_bf16_kernel<<<(int)((tot4 + 255) / 256), 256, 0, stream>>>(
        encoder_out, sort_ind, enc_sb);
  }

  // K3: enc_feat GEMM ([25088,2048] x [2048,512]) -> bf16, bias folded
  {
    int waves = ((BB * PP) / 16) * (AA / 16);   // 50176 wave-tiles
    enc_feat_kernel<<<(waves + 7) / 8, 256, 0, stream>>>(encoder_out, sort_ind,
                                                         wenc, enc_attn_b, enc_featb);
  }

  // K4: h0 / c0
  init_state_kernel<<<64, 256, 0, stream>>>(mean_b, winh, winc, init_h_b, init_c_b,
                                            h_f, c_f, h_b);

  // K5: persistent 49-step decoder (8 blocks x 16 batch rows) — recurrence only
  decoder_step_kernel<<<BB / 16, 512, 0, stream>>>(
      encoder_out, enc_sb, use_bf16_enc, sort_ind, dec_len_s, enc_featb, emb_b,
      wdec, dec_attn_b, full_attn_W, full_attn_b,
      wfbeta, fbeta_b, wih, whh, lstm_bih, lstm_bhh,
      h_f, c_f, h_b, ctx_f, xctx_b, gates_f,
      h_hist, out_alphas);

  // K6: hoisted fc GEMM over all (b,t) rows at full occupancy
  {
    int waves = ((BB * TT) / 16) * (VV / 16);   // 245,000 wave-tiles
    fc_logits_kernel<<<(waves + 7) / 8, 256, 0, stream>>>(
        h_hist, wfc, fc_b, dec_len_s, out_logits);
  }
}
